// LongformerMaxp_42348377539225
// MI455X (gfx1250) — matrix operand order repaired and checked by reading, not compiled, not run on hardware
//
#include <hip/hip_runtime.h>
#include <hip/hip_bf16.h>

// ---------------- model constants ----------------
#define LAY   4
#define NHEAD 12
#define HID   768
#define DFF   3072
#define WSZ   256
#define SEQ   1536
#define GN    64
#define NUMB  2
#define BATCH (NUMB*3)          // 6 sequences
#define TOK   (BATCH*SEQ)       // 9216 rows
#define DH    (HID/NHEAD)       // 64
#define HH    (HID*HID)
#define HD    (HID*DFF)

typedef __attribute__((ext_vector_type(16))) _Float16 v16h;
typedef __attribute__((ext_vector_type(8)))  float    v8f;

// gfx1250 async global->LDS path (probe via __has_builtin; clean fallback)
#if defined(__gfx1250__) && __has_builtin(__builtin_amdgcn_global_load_async_to_lds_b128) && __has_builtin(__builtin_amdgcn_s_wait_asynccnt)
#define USE_ASYNC_LDS 1
// builtin signature (from compiler diagnostic): V4i __device__* src, V4i __shared__* dst, imm offset, imm cpol
typedef int v4i __attribute__((vector_size(16)));
typedef __attribute__((address_space(1))) v4i gv4i;   // global (AS1)
typedef __attribute__((address_space(3))) v4i sv4i;   // LDS (AS3)
#define ASYNC_B128(srcp, dstp) \
  __builtin_amdgcn_global_load_async_to_lds_b128((gv4i*)(srcp), (sv4i*)(dstp), 0, 0)
#else
#define USE_ASYNC_LDS 0
#endif

// ---------------- weight convert + transpose: f32 [K,N] -> f16 [N,K] ----------------
__global__ void convt_kernel(const float* __restrict__ W, _Float16* __restrict__ out,
                             int K, int N) {
  size_t idx = (size_t)blockIdx.x * 256 + threadIdx.x;
  size_t tot = (size_t)K * N;
  if (idx >= tot) return;
  size_t n = idx / K, k = idx % K;
  out[idx] = (_Float16)W[k * (size_t)N + n];
}

// ---------------- WMMA GEMM: C[M,N] = act(((A[M,K] @ Wt^T) + bias) * scale) ----------------
// Wt is f16 [N,K] row-major (pre-transposed weight). act: 0=none 1=gelu
// Double-buffered LDS; B tile staged via async-to-LDS when available.
#define TM 128
#define TN 128
#define TK 32
#define LSTR (TK + 16)   // 48 halves -> 96B row stride, keeps 16B alignment

__global__ __launch_bounds__(256) void gemm_wmma_kernel(
    const float* __restrict__ A, const _Float16* __restrict__ Bt,
    const float* __restrict__ bias, float* __restrict__ C,
    int M, int N, int K, int act, float scale)
{
  __shared__ _Float16 lA[2][TM][LSTR];
  __shared__ _Float16 lB[2][TN][LSTR];

  const int tid  = threadIdx.x;
  const int lane = tid & 31;
  const int wave = tid >> 5;
  const int wm   = wave >> 1;        // 0..3 -> 32-row slab
  const int wn   = wave & 1;         // 0..1 -> 64-col slab
  const int half = lane >> 4;        // hi/lo 16 lanes
  const int l16  = lane & 15;
  const int tileM = blockIdx.y * TM;
  const int tileN = blockIdx.x * TN;

  v8f acc[2][4];
  const v8f vzero = {0.f,0.f,0.f,0.f,0.f,0.f,0.f,0.f};
#pragma unroll
  for (int mi = 0; mi < 2; ++mi)
#pragma unroll
    for (int ni = 0; ni < 4; ++ni) acc[mi][ni] = vzero;

  const int rowS = tid >> 1;           // 0..127
  const int colS = (tid & 1) << 4;     // 0 or 16

  const float*    gaBase = A  + (size_t)(tileM + rowS) * K;
  const _Float16* gbBase = Bt + (size_t)(tileN + rowS) * K;

  // ---- prologue: stage K-tile 0 into buffer 0 ----
  {
    const float* ga = gaBase + colS;
    float4 f0 = ((const float4*)ga)[0];
    float4 f1 = ((const float4*)ga)[1];
    float4 f2 = ((const float4*)ga)[2];
    float4 f3 = ((const float4*)ga)[3];
    union { _Float16 h[16]; uint4 u[2]; } hb;
    hb.h[0]=(_Float16)f0.x; hb.h[1]=(_Float16)f0.y; hb.h[2]=(_Float16)f0.z; hb.h[3]=(_Float16)f0.w;
    hb.h[4]=(_Float16)f1.x; hb.h[5]=(_Float16)f1.y; hb.h[6]=(_Float16)f1.z; hb.h[7]=(_Float16)f1.w;
    hb.h[8]=(_Float16)f2.x; hb.h[9]=(_Float16)f2.y; hb.h[10]=(_Float16)f2.z; hb.h[11]=(_Float16)f2.w;
    hb.h[12]=(_Float16)f3.x; hb.h[13]=(_Float16)f3.y; hb.h[14]=(_Float16)f3.z; hb.h[15]=(_Float16)f3.w;
    *(uint4*)&lA[0][rowS][colS]     = hb.u[0];
    *(uint4*)&lA[0][rowS][colS + 8] = hb.u[1];

    const _Float16* gb = gbBase + colS;
#if USE_ASYNC_LDS
    ASYNC_B128(gb,     &lB[0][rowS][colS]);
    ASYNC_B128(gb + 8, &lB[0][rowS][colS + 8]);
#else
    uint4 b0 = ((const uint4*)gb)[0];
    uint4 b1 = ((const uint4*)gb)[1];
    *(uint4*)&lB[0][rowS][colS]     = b0;
    *(uint4*)&lB[0][rowS][colS + 8] = b1;
#endif
  }
#if USE_ASYNC_LDS
  __builtin_amdgcn_s_wait_asynccnt(0);
#endif
  __syncthreads();

  const int KT = K / TK;
  for (int kt = 0; kt < KT; ++kt) {
    const int cur = kt & 1, nxt = cur ^ 1;
    const bool more = (kt + 1 < KT);
    const int knext = (kt + 1) * TK;

    // ---- 1) issue global fetches for next tile (overlap with WMMA below) ----
    float4 f0 = {}, f1 = {}, f2 = {}, f3 = {};
    if (more) {
      const float* ga = gaBase + knext + colS;
      f0 = ((const float4*)ga)[0];
      f1 = ((const float4*)ga)[1];
      f2 = ((const float4*)ga)[2];
      f3 = ((const float4*)ga)[3];
#if USE_ASYNC_LDS
      const _Float16* gb = gbBase + knext + colS;
      ASYNC_B128(gb,     &lB[nxt][rowS][colS]);
      ASYNC_B128(gb + 8, &lB[nxt][rowS][colS + 8]);
#endif
      if (kt + 2 < KT) {
        __builtin_prefetch(gaBase + knext + TK + colS, 0, 1);
        __builtin_prefetch(gbBase + knext + TK + colS, 0, 1);
      }
    }
#if !USE_ASYNC_LDS
    uint4 b0 = {}, b1 = {};
    if (more) {
      const _Float16* gb = gbBase + knext + colS;
      b0 = ((const uint4*)gb)[0];
      b1 = ((const uint4*)gb)[1];
    }
#endif

    // ---- 2) build fragments per ISA 7.12.2 lane layouts, run 8 WMMAs ----
    v16h afr[2], bfr[4];
#pragma unroll
    for (int mi = 0; mi < 2; ++mi) {
      int r = wm * 32 + mi * 16 + l16;
      union { uint4 u[2]; v16h v; } t;
      t.u[0] = *(const uint4*)&lA[cur][r][half * 8];        // K = h*8 .. h*8+7
      t.u[1] = *(const uint4*)&lA[cur][r][16 + half * 8];   // K = 16+h*8 .. 16+h*8+7
      afr[mi] = t.v;
    }
#pragma unroll
    for (int ni = 0; ni < 4; ++ni) {
      int c = wn * 64 + ni * 16 + l16;
      union { uint4 u[2]; v16h v; } t;
      t.u[0] = *(const uint4*)&lB[cur][c][half * 8];        // K = h*8 .. h*8+7
      t.u[1] = *(const uint4*)&lB[cur][c][16 + half * 8];   // K = 16+h*8 .. 16+h*8+7
      bfr[ni] = t.v;
    }
#pragma unroll
    for (int mi = 0; mi < 2; ++mi)
#pragma unroll
      for (int ni = 0; ni < 4; ++ni)
        acc[mi][ni] = __builtin_amdgcn_wmma_f32_16x16x32_f16(
            false, afr[mi], false, bfr[ni], (short)0, acc[mi][ni], false, false);

    // ---- 3) convert & park next A tile (and B on fallback path) ----
    if (more) {
      union { _Float16 h[16]; uint4 u[2]; } hb;
      hb.h[0]=(_Float16)f0.x; hb.h[1]=(_Float16)f0.y; hb.h[2]=(_Float16)f0.z; hb.h[3]=(_Float16)f0.w;
      hb.h[4]=(_Float16)f1.x; hb.h[5]=(_Float16)f1.y; hb.h[6]=(_Float16)f1.z; hb.h[7]=(_Float16)f1.w;
      hb.h[8]=(_Float16)f2.x; hb.h[9]=(_Float16)f2.y; hb.h[10]=(_Float16)f2.z; hb.h[11]=(_Float16)f2.w;
      hb.h[12]=(_Float16)f3.x; hb.h[13]=(_Float16)f3.y; hb.h[14]=(_Float16)f3.z; hb.h[15]=(_Float16)f3.w;
      *(uint4*)&lA[nxt][rowS][colS]     = hb.u[0];
      *(uint4*)&lA[nxt][rowS][colS + 8] = hb.u[1];
#if !USE_ASYNC_LDS
      *(uint4*)&lB[nxt][rowS][colS]     = b0;
      *(uint4*)&lB[nxt][rowS][colS + 8] = b1;
#endif
    }
#if USE_ASYNC_LDS
    __builtin_amdgcn_s_wait_asynccnt(0);
#endif
    __syncthreads();
  }

  // ---- epilogue: bias, scale, activation, store fp32 ----
#pragma unroll
  for (int mi = 0; mi < 2; ++mi)
#pragma unroll
    for (int ni = 0; ni < 4; ++ni) {
      int col = tileN + wn * 64 + ni * 16 + l16;
      float bv = bias[col];
#pragma unroll
      for (int r = 0; r < 8; ++r) {
        int row = tileM + wm * 32 + mi * 16 + half * 8 + r;
        float v = (acc[mi][ni][r] + bv) * scale;
        if (act == 1) {
          float u = 0.7978845608028654f * (v + 0.044715f * v * v * v);
          v = 0.5f * v * (1.0f + tanhf(u));
        }
        C[(size_t)row * N + col] = v;
      }
    }
}

// ---------------- embedding gather ----------------
__global__ void embed_kernel(const int* __restrict__ ids, const float* __restrict__ wemb,
                             const float* __restrict__ pemb, float* __restrict__ out) {
  int row = blockIdx.x;               // token index 0..TOK-1
  int c   = threadIdx.x;
  int id  = ids[row];
  int sp  = row % SEQ;
  for (int i = 0; i < HID; i += 256)
    out[(size_t)row * HID + c + i] =
        wemb[(size_t)id * HID + c + i] + pemb[(size_t)sp * HID + c + i];
}

// ---------------- LayerNorm (wave-per-row), optional residual ----------------
__global__ __launch_bounds__(256) void ln_kernel(
    const float* __restrict__ x, const float* __restrict__ res,
    const float* __restrict__ sc, const float* __restrict__ bi,
    float* __restrict__ out, int rows, int has_res)
{
  int wid  = (blockIdx.x * blockDim.x + threadIdx.x) >> 5;
  int lane = threadIdx.x & 31;
  if (wid >= rows) return;
  const float* xr = x   + (size_t)wid * HID;
  const float* rr = res + (size_t)wid * HID;
  float vals[HID / 32];
  float sum = 0.f;
  for (int i = 0; i < HID / 32; ++i) {
    float t = xr[lane + i * 32];
    if (has_res) t += rr[lane + i * 32];
    vals[i] = t; sum += t;
  }
  for (int off = 16; off; off >>= 1) sum += __shfl_xor(sum, off, 32);
  float mean = sum * (1.0f / HID);
  float var = 0.f;
  for (int i = 0; i < HID / 32; ++i) { float d = vals[i] - mean; var += d * d; }
  for (int off = 16; off; off >>= 1) var += __shfl_xor(var, off, 32);
  var *= (1.0f / HID);
  float inv = rsqrtf(var + 1e-5f);
  for (int i = 0; i < HID / 32; ++i) {
    int c = lane + i * 32;
    out[(size_t)wid * HID + c] = (vals[i] - mean) * inv * sc[c] + bi[c];
  }
}

// ---------------- banded local attention, wave-per-query, online softmax ----------------
__device__ __forceinline__ void attn_step(float s, const float* vp, int lane,
                                          float& m, float& l, float& o0, float& o1) {
  float mn  = fmaxf(m, s);
  float cor = __expf(m - mn);
  float p   = __expf(s - mn);
  l = l * cor + p;
  o0 = o0 * cor + p * vp[lane];
  o1 = o1 * cor + p * vp[lane + 32];
  m = mn;
}

__global__ __launch_bounds__(256) void attn_band_kernel(
    const float* __restrict__ q, const float* __restrict__ k, const float* __restrict__ v,
    const int* __restrict__ mask, float* __restrict__ out)
{
  int wid  = (blockIdx.x * blockDim.x + threadIdx.x) >> 5;
  int lane = threadIdx.x & 31;
  if (wid >= BATCH * NHEAD * SEQ) return;
  int qpos = wid % SEQ; int t = wid / SEQ;
  int hh   = t % NHEAD; int b = t / NHEAD;
  const float* qp = q + ((size_t)b * SEQ + qpos) * HID + hh * DH;
  float q0 = qp[lane], q1 = qp[lane + 32];
  const int* mrow = mask + b * SEQ;
  float m = -1e30f, l = 0.f, o0 = 0.f, o1 = 0.f;

  // global keys 0..GN-1 (attended by every query, scg path)
  for (int j = 0; j < GN; ++j) {
    const float* kp = k + ((size_t)b * SEQ + j) * HID + hh * DH;
    float part = q0 * kp[lane] + q1 * kp[lane + 32];
    for (int off = 16; off; off >>= 1) part += __shfl_xor(part, off, 32);
    float s = (mrow[j] > 0) ? part : -1e9f;
    attn_step(s, v + ((size_t)b * SEQ + j) * HID + hh * DH, lane, m, l, o0, o1);
  }
  // band keys: |j - qpos| <= WSZ, j >= GN, j < SEQ
  int jlo = qpos - WSZ; if (jlo < GN) jlo = GN;
  int jhi = qpos + WSZ; if (jhi > SEQ - 1) jhi = SEQ - 1;
  for (int j = jlo; j <= jhi; ++j) {
    const float* kp = k + ((size_t)b * SEQ + j) * HID + hh * DH;
    float part = q0 * kp[lane] + q1 * kp[lane + 32];
    for (int off = 16; off; off >>= 1) part += __shfl_xor(part, off, 32);
    float s = (mrow[j] > 0) ? part : -1e9f;
    attn_step(s, v + ((size_t)b * SEQ + j) * HID + hh * DH, lane, m, l, o0, o1);
  }
  float inv = 1.0f / l;
  float* op = out + ((size_t)b * SEQ + qpos) * HID + hh * DH;
  op[lane] = o0 * inv; op[lane + 32] = o1 * inv;
}

// ---------------- global attention: overwrites rows qpos < GN ----------------
__global__ __launch_bounds__(256) void attn_global_kernel(
    const float* __restrict__ qg, const float* __restrict__ kg, const float* __restrict__ vg,
    const int* __restrict__ mask, float* __restrict__ out)
{
  int wid  = (blockIdx.x * blockDim.x + threadIdx.x) >> 5;
  int lane = threadIdx.x & 31;
  if (wid >= BATCH * NHEAD * GN) return;
  int qpos = wid % GN; int t = wid / GN;
  int hh   = t % NHEAD; int b = t / NHEAD;
  const float* qp = qg + ((size_t)b * SEQ + qpos) * HID + hh * DH;
  float q0 = qp[lane], q1 = qp[lane + 32];
  const int* mrow = mask + b * SEQ;
  float m = -1e30f, l = 0.f, o0 = 0.f, o1 = 0.f;
  for (int j = 0; j < SEQ; ++j) {
    const float* kp = kg + ((size_t)b * SEQ + j) * HID + hh * DH;
    float part = q0 * kp[lane] + q1 * kp[lane + 32];
    for (int off = 16; off; off >>= 1) part += __shfl_xor(part, off, 32);
    float s = (mrow[j] > 0) ? part : -1e9f;
    attn_step(s, vg + ((size_t)b * SEQ + j) * HID + hh * DH, lane, m, l, o0, o1);
  }
  float inv = 1.0f / l;
  float* op = out + ((size_t)b * SEQ + qpos) * HID + hh * DH;
  op[lane] = o0 * inv; op[lane + 32] = o1 * inv;
}

// ---------------- pooling + classifier head ----------------
__global__ void cls_max_kernel(const float* __restrict__ h, float* __restrict__ cls) {
  int idx = blockIdx.x * blockDim.x + threadIdx.x;
  if (idx >= NUMB * HID) return;
  int n = idx / HID, c = idx % HID;
  float m = -3.0e38f;
  for (int r = 0; r < 3; ++r)
    m = fmaxf(m, h[((size_t)(n * 3 + r) * SEQ) * HID + c]);
  cls[idx] = m;
}

__global__ void dense_kernel(const float* __restrict__ cls, const float* __restrict__ W,
                             const float* __restrict__ b, float* __restrict__ hs) {
  int idx = blockIdx.x * blockDim.x + threadIdx.x;
  if (idx >= NUMB * HID) return;
  int n = idx / HID, c = idx % HID;
  float acc = b[c];
  for (int k = 0; k < HID; ++k) acc += cls[n * HID + k] * W[(size_t)k * HID + c];
  hs[idx] = tanhf(acc);
}

__global__ void out_kernel(const float* __restrict__ hs, const float* __restrict__ W,
                           const float* __restrict__ b, float* __restrict__ out) {
  int idx = threadIdx.x;
  if (idx >= NUMB * 2) return;
  int n = idx / 2, j = idx % 2;
  float acc = b[j];
  for (int k = 0; k < HID; ++k) acc += hs[n * HID + k] * W[k * 2 + j];
  out[idx] = acc;                 // score (== logits flat)
  out[NUMB * 2 + idx] = acc;      // logits
}

// ---------------- host orchestration ----------------
static inline size_t al256(size_t x) { return (x + 255) & ~(size_t)255; }

extern "C" void kernel_launch(void* const* d_in, const int* in_sizes, int n_in,
                              void* d_out, int out_size, void* d_ws, size_t ws_size,
                              hipStream_t stream) {
  (void)in_sizes; (void)n_in; (void)out_size; (void)ws_size;
  const int*   ids     = (const int*)  d_in[0];
  const int*   maskp   = (const int*)  d_in[1];
  const float* wemb    = (const float*)d_in[4];
  const float* pemb    = (const float*)d_in[5];
  const float* emb_s   = (const float*)d_in[6];
  const float* emb_b   = (const float*)d_in[7];
  const float* Wq  = (const float*)d_in[8];
  const float* Wk  = (const float*)d_in[9];
  const float* Wv  = (const float*)d_in[10];
  const float* Wo  = (const float*)d_in[11];
  const float* Wqg = (const float*)d_in[12];
  const float* Wkg = (const float*)d_in[13];
  const float* Wvg = (const float*)d_in[14];
  const float* bq  = (const float*)d_in[15];
  const float* bk  = (const float*)d_in[16];
  const float* bv  = (const float*)d_in[17];
  const float* bo  = (const float*)d_in[18];
  const float* bqg = (const float*)d_in[19];
  const float* bkg = (const float*)d_in[20];
  const float* bvg = (const float*)d_in[21];
  const float* W1  = (const float*)d_in[22];
  const float* b1  = (const float*)d_in[23];
  const float* W2  = (const float*)d_in[24];
  const float* b2  = (const float*)d_in[25];
  const float* ln1s = (const float*)d_in[26];
  const float* ln1b = (const float*)d_in[27];
  const float* ln2s = (const float*)d_in[28];
  const float* ln2b = (const float*)d_in[29];
  const float* denseW = (const float*)d_in[30];
  const float* denseb = (const float*)d_in[31];
  const float* outW   = (const float*)d_in[32];
  const float* outb   = (const float*)d_in[33];

  // workspace carve
  char* w = (char*)d_ws;
  size_t off = 0;
  auto carve = [&](size_t bytes) { void* p = w + off; off = al256(off + bytes); return p; };
  _Float16* wqT  = (_Float16*)carve((size_t)LAY * HH * 2);
  _Float16* wkT  = (_Float16*)carve((size_t)LAY * HH * 2);
  _Float16* wvT  = (_Float16*)carve((size_t)LAY * HH * 2);
  _Float16* woT  = (_Float16*)carve((size_t)LAY * HH * 2);
  _Float16* wqgT = (_Float16*)carve((size_t)LAY * HH * 2);
  _Float16* wkgT = (_Float16*)carve((size_t)LAY * HH * 2);
  _Float16* wvgT = (_Float16*)carve((size_t)LAY * HH * 2);
  _Float16* w1T  = (_Float16*)carve((size_t)LAY * HD * 2);
  _Float16* w2T  = (_Float16*)carve((size_t)LAY * HD * 2);
  float* h    = (float*)carve((size_t)TOK * HID * 4);
  float* qb   = (float*)carve((size_t)TOK * HID * 4);
  float* kb   = (float*)carve((size_t)TOK * HID * 4);
  float* vb   = (float*)carve((size_t)TOK * HID * 4);
  float* qgb  = (float*)carve((size_t)TOK * HID * 4);
  float* kgb  = (float*)carve((size_t)TOK * HID * 4);
  float* vgb  = (float*)carve((size_t)TOK * HID * 4);
  float* attn = (float*)carve((size_t)TOK * HID * 4);
  float* tmp  = (float*)carve((size_t)TOK * HID * 4);
  float* mid  = (float*)carve((size_t)TOK * DFF * 4);
  float* cls  = (float*)carve((size_t)NUMB * HID * 4);
  float* hs   = (float*)carve((size_t)NUMB * HID * 4);

  auto convt = [&](const float* src, _Float16* dst, int K, int N) {
    size_t tot = (size_t)K * N;
    convt_kernel<<<dim3((unsigned)((tot + 255) / 256)), dim3(256), 0, stream>>>(src, dst, K, N);
  };
  auto gemm = [&](const float* A, const _Float16* Bt, const float* bias, float* C,
                  int M, int N, int K, int act, float scale) {
    dim3 g(N / TN, M / TM);
    gemm_wmma_kernel<<<g, dim3(256), 0, stream>>>(A, Bt, bias, C, M, N, K, act, scale);
  };

  // weight conversion (deterministic every call)
  for (int l = 0; l < LAY; ++l) {
    convt(Wq  + (size_t)l * HH, wqT  + (size_t)l * HH, HID, HID);
    convt(Wk  + (size_t)l * HH, wkT  + (size_t)l * HH, HID, HID);
    convt(Wv  + (size_t)l * HH, wvT  + (size_t)l * HH, HID, HID);
    convt(Wo  + (size_t)l * HH, woT  + (size_t)l * HH, HID, HID);
    convt(Wqg + (size_t)l * HH, wqgT + (size_t)l * HH, HID, HID);
    convt(Wkg + (size_t)l * HH, wkgT + (size_t)l * HH, HID, HID);
    convt(Wvg + (size_t)l * HH, wvgT + (size_t)l * HH, HID, HID);
    convt(W1  + (size_t)l * HD, w1T  + (size_t)l * HD, HID, DFF);
    convt(W2  + (size_t)l * HD, w2T  + (size_t)l * HD, DFF, HID);
  }

  // embedding + LN
  embed_kernel<<<dim3(TOK), dim3(256), 0, stream>>>(ids, wemb, pemb, tmp);
  ln_kernel<<<dim3(TOK / 8), dim3(256), 0, stream>>>(tmp, tmp, emb_s, emb_b, h, TOK, 0);

  const float scale = 0.125f;   // 1/sqrt(64)
  for (int l = 0; l < LAY; ++l) {
    gemm(h, wqT  + (size_t)l * HH, bq  + l * HID, qb,  TOK, HID, HID, 0, scale);
    gemm(h, wkT  + (size_t)l * HH, bk  + l * HID, kb,  TOK, HID, HID, 0, 1.f);
    gemm(h, wvT  + (size_t)l * HH, bv  + l * HID, vb,  TOK, HID, HID, 0, 1.f);
    gemm(h, wqgT + (size_t)l * HH, bqg + l * HID, qgb, TOK, HID, HID, 0, scale);
    gemm(h, wkgT + (size_t)l * HH, bkg + l * HID, kgb, TOK, HID, HID, 0, 1.f);
    gemm(h, wvgT + (size_t)l * HH, bvg + l * HID, vgb, TOK, HID, HID, 0, 1.f);

    attn_band_kernel<<<dim3((BATCH * NHEAD * SEQ) / 8), dim3(256), 0, stream>>>(
        qb, kb, vb, maskp, attn);
    attn_global_kernel<<<dim3((BATCH * NHEAD * GN) / 8), dim3(256), 0, stream>>>(
        qgb, kgb, vgb, maskp, attn);

    gemm(attn, woT + (size_t)l * HH, bo + l * HID, tmp, TOK, HID, HID, 0, 1.f);
    ln_kernel<<<dim3(TOK / 8), dim3(256), 0, stream>>>(h, tmp, ln1s + l * HID, ln1b + l * HID, h, TOK, 1);

    gemm(h,   w1T + (size_t)l * HD, b1 + l * DFF, mid, TOK, DFF, HID, 1, 1.f);
    gemm(mid, w2T + (size_t)l * HD, b2 + l * HID, tmp, TOK, HID, DFF, 0, 1.f);
    ln_kernel<<<dim3(TOK / 8), dim3(256), 0, stream>>>(h, tmp, ln2s + l * HID, ln2b + l * HID, h, TOK, 1);
  }

  cls_max_kernel<<<dim3((NUMB * HID + 255) / 256), dim3(256), 0, stream>>>(h, cls);
  dense_kernel<<<dim3((NUMB * HID + 255) / 256), dim3(256), 0, stream>>>(cls, denseW, denseb, hs);
  out_kernel<<<dim3(1), dim3(64), 0, stream>>>(hs, outW, outb, (float*)d_out);
}
